// SelfAttention_8873402434238
// MI455X (gfx1250) — compile-verified, matrix operand order checked by reading, not run on hardware
//
#include <hip/hip_runtime.h>
#include <hip/hip_bf16.h>
#include <math.h>

typedef __bf16 bf16_t;
typedef __attribute__((ext_vector_type(8)))  bf16_t v8bf;
typedef __attribute__((ext_vector_type(16))) bf16_t v16bf;
typedef __attribute__((ext_vector_type(8)))  float  v8f;
typedef __attribute__((ext_vector_type(4)))  int    v4i;

#define AS_GLOBAL __attribute__((address_space(1)))
#define AS_LDS    __attribute__((address_space(3)))

constexpr int BB = 2, TT = 4096, CC = 768, HH = 12, DD = 64;
constexpr float SCALE = 0.03608439182435161f; // 1/sqrt(768): reference scales by 1/sqrt(C)

// Optional CDNA5 async global->LDS path (guarded; falls back to sync staging).
#if defined(__has_builtin)
#if __has_builtin(__builtin_amdgcn_global_load_async_to_lds_b128) && \
    __has_builtin(__builtin_amdgcn_s_wait_asynccnt)
#define USE_ASYNC_LDS 1
#endif
#endif
#ifndef USE_ASYNC_LDS
#define USE_ASYNC_LDS 0
#endif

static __device__ __forceinline__ bf16_t f2bf(float f) {
  unsigned u = __float_as_uint(f);
  u += 0x7FFFu + ((u >> 16) & 1u);            // round-to-nearest-even
  unsigned short s = (unsigned short)(u >> 16);
  return __builtin_bit_cast(bf16_t, s);
}

static __device__ __forceinline__ v8bf ld8(const bf16_t* p) {
  return *reinterpret_cast<const v8bf*>(p);
}

static __device__ __forceinline__ v16bf joinbf(v8bf a, v8bf b) {
  return __builtin_shufflevector(a, b, 0,1,2,3,4,5,6,7,8,9,10,11,12,13,14,15);
}

// Load one 16-lane-striped bf16 A/B fragment covering a 32-wide K chunk.
// ISA layout: lanes 0-15 hold K {0..7,16..23}, lanes 16-31 hold K {8..15,24..31}.
static __device__ __forceinline__ v16bf frag_ld(const bf16_t* rowbase, int half) {
  v8bf lo = ld8(rowbase + half * 8);
  v8bf hi = ld8(rowbase + 16 + half * 8);
  return joinbf(lo, hi);
}

static __device__ __forceinline__ v8f wmma_bf16(v16bf a, v16bf b, v8f c) {
  return __builtin_amdgcn_wmma_f32_16x16x32_bf16(false, a, false, b, (short)0, c,
                                                 false, false);
}

// ---------------------------------------------------------------- convert ---
__global__ void cvt_f32_bf16(const float* __restrict__ in, bf16_t* __restrict__ out,
                             int n) {
  int i = blockIdx.x * blockDim.x + threadIdx.x;
  int stride = gridDim.x * blockDim.x;
  for (; i < n; i += stride) out[i] = f2bf(in[i]);
}

// ------------------------------------------------------------------- GEMM ---
// C[M,N] = A[M,K] * Bw[K,N], bf16 in, fp32 accumulate. 128x128 block tile,
// 8 waves in a 4x2 grid, 32x64 per wave, BK=32 (one bf16 WMMA K-step).
// MODE 0: store bf16 to [B,H,T,D]   (Q, K)
// MODE 1: store bf16 to [B,H,D,T]   (V transposed; contiguous-t packed stores)
// MODE 2: store fp32 row-major [M,N] (final output)
template <int MODE>
__global__ __launch_bounds__(256)
void gemm_bf16(const bf16_t* __restrict__ A, const bf16_t* __restrict__ Bw,
               void* __restrict__ out, int M, int N, int K) {
  __shared__ bf16_t Al[128][32];   // A tile, row-major
  __shared__ bf16_t Btl[128][32];  // B tile stored transposed: [n][k]

  const int tid = threadIdx.x;
  const int lane = tid & 31, wave = tid >> 5;
  const int half = (lane >> 4) & 1, l15 = lane & 15;
  const int wm = wave >> 1, wn = wave & 1;        // 4x2 wave grid
  const int m0 = blockIdx.y * 128, n0 = blockIdx.x * 128;

  v8f acc[2][4];
#pragma unroll
  for (int i = 0; i < 2; ++i)
#pragma unroll
    for (int j = 0; j < 4; ++j) acc[i][j] = (v8f){0.f,0.f,0.f,0.f,0.f,0.f,0.f,0.f};

  const int arow = tid >> 1, acol = (tid & 1) * 16;   // 128x32 A tile
  const int brow = tid >> 3, bcol = (tid & 7) * 16;   // 32x128 B tile
  const bf16_t* aptr = &A[(size_t)(m0 + arow) * K + acol];

  for (int k0 = 0; k0 < K; k0 += 32) {
    if (k0 + 32 < K) __builtin_prefetch(aptr + k0 + 32, 0, 0);  // next A tile
    v8bf b0 = ld8(&Bw[(size_t)(k0 + brow) * N + n0 + bcol]);
    v8bf b1 = ld8(&Bw[(size_t)(k0 + brow) * N + n0 + bcol + 8]);
#if !USE_ASYNC_LDS
    v8bf a0 = ld8(aptr + k0);
    v8bf a1 = ld8(aptr + k0 + 8);
#endif
    __syncthreads();  // previous tile fully consumed before LDS is rewritten
#if USE_ASYNC_LDS
    __builtin_amdgcn_global_load_async_to_lds_b128(
        (AS_GLOBAL v4i*)(aptr + k0),
        (AS_LDS v4i*)&Al[arow][acol], 0, 0);
    __builtin_amdgcn_global_load_async_to_lds_b128(
        (AS_GLOBAL v4i*)(aptr + k0 + 8),
        (AS_LDS v4i*)&Al[arow][acol + 8], 0, 0);
#else
    *reinterpret_cast<v8bf*>(&Al[arow][acol]) = a0;
    *reinterpret_cast<v8bf*>(&Al[arow][acol + 8]) = a1;
#endif
#pragma unroll
    for (int j = 0; j < 8; ++j) {
      Btl[bcol + j][brow] = b0[j];
      Btl[bcol + 8 + j][brow] = b1[j];
    }
#if USE_ASYNC_LDS
    __builtin_amdgcn_s_wait_asynccnt(0);
#endif
    __syncthreads();

    v16bf af[2];
#pragma unroll
    for (int i = 0; i < 2; ++i)
      af[i] = frag_ld(&Al[wm * 32 + i * 16 + l15][0], half);
#pragma unroll
    for (int j = 0; j < 4; ++j) {
      v16bf bfj = frag_ld(&Btl[wn * 64 + j * 16 + l15][0], half);
#pragma unroll
      for (int i = 0; i < 2; ++i) acc[i][j] = wmma_bf16(af[i], bfj, acc[i][j]);
    }
  }

  // Epilogue. C/D layout: slot s -> row (s or s+8 by half), col = l15.
#pragma unroll
  for (int i = 0; i < 2; ++i)
#pragma unroll
    for (int j = 0; j < 4; ++j) {
      const int n = n0 + wn * 64 + j * 16 + l15;
      const int mb = m0 + wm * 32 + i * 16 + 8 * half;  // slot-0 row
      if (MODE == 2) {
#pragma unroll
        for (int s = 0; s < 8; ++s)
          ((float*)out)[(size_t)(mb + s) * N + n] = acc[i][j][s];
      } else if (MODE == 0) {
        const int h = n >> 6, d = n & (DD - 1);
#pragma unroll
        for (int s = 0; s < 8; ++s) {
          int m = mb + s;
          int b = m >> 12, t = m & (TT - 1);
          ((bf16_t*)out)[(((size_t)(b * HH + h)) * TT + t) * DD + d] =
              f2bf(acc[i][j][s]);
        }
      } else {  // MODE 1: [B,H,D,T]; 8 slots are contiguous in t -> one b128
        const int h = n >> 6, d = n & (DD - 1);
        const int b = mb >> 12, t = mb & (TT - 1);
        v8bf pk;
#pragma unroll
        for (int s = 0; s < 8; ++s) pk[s] = f2bf(acc[i][j][s]);
        *reinterpret_cast<v8bf*>(
            &((bf16_t*)out)[(((size_t)(b * HH + h)) * DD + d) * TT + t]) = pk;
      }
    }
}

// -------------------------------------------------------- flash attention ---
// Grid: (T/64, B*H). Block: 128 threads = 4 waves; each wave owns 16 Q rows.
__global__ __launch_bounds__(128)
void flash_attn(const bf16_t* __restrict__ Q, const bf16_t* __restrict__ Kk,
                const bf16_t* __restrict__ Vt, bf16_t* __restrict__ O) {
  __shared__ bf16_t Pl[4][16][64];  // per-wave P staging (C-layout -> A-layout)

  const int tid = threadIdx.x, lane = tid & 31, w = tid >> 5;
  const int half = (lane >> 4) & 1, l15 = lane & 15;
  const int qb = blockIdx.x, bh = blockIdx.y;
  const int t0 = qb * 64, trow0 = t0 + w * 16;
  const size_t baseQK = (size_t)bh * TT * DD;
  const size_t baseVt = (size_t)bh * DD * TT;

  // Q fragments held in registers for the whole kernel (row = trow0 + l15).
  const bf16_t* qrow = Q + baseQK + (size_t)(trow0 + l15) * DD;
  v16bf qf0 = frag_ld(qrow, half);        // d 0..31
  v16bf qf1 = frag_ld(qrow + 32, half);   // d 32..63

  v8f o[4];
  float m_[8], l_[8];
#pragma unroll
  for (int dt = 0; dt < 4; ++dt) o[dt] = (v8f){0.f,0.f,0.f,0.f,0.f,0.f,0.f,0.f};
#pragma unroll
  for (int i = 0; i < 8; ++i) { m_[i] = -1e30f; l_[i] = 0.f; }

  for (int sb = 0; sb <= qb; ++sb) {
    const int s0 = sb * 64;
    const bool diag = (sb == qb);
    if (!diag) {  // streaming hint for the next K/V block
      __builtin_prefetch(Kk + baseQK + (size_t)(s0 + 64 + (lane & 63)) * DD, 0, 0);
      __builtin_prefetch(Vt + baseVt + (size_t)lane * TT + s0 + 64, 0, 0);
    }

    // S = Q * K^T  (B-fragment column = key index s; contiguous d loads)
    v8f sacc[4];
#pragma unroll
    for (int ct = 0; ct < 4; ++ct) {
      const bf16_t* krow = Kk + baseQK + (size_t)(s0 + ct * 16 + l15) * DD;
      v16bf kf0 = frag_ld(krow, half);
      v16bf kf1 = frag_ld(krow + 32, half);
      v8f z = (v8f){0.f,0.f,0.f,0.f,0.f,0.f,0.f,0.f};
      z = wmma_bf16(qf0, kf0, z);
      z = wmma_bf16(qf1, kf1, z);
      sacc[ct] = z;
    }

    // Scale + causal mask (finite -inf to avoid inf-inf NaN in the max path)
    float p[4][8];
#pragma unroll
    for (int ct = 0; ct < 4; ++ct) {
      int col = s0 + ct * 16 + l15;
#pragma unroll
      for (int i = 0; i < 8; ++i) {
        float v = sacc[ct][i] * SCALE;
        int row = trow0 + i + 8 * half;
        if (diag && col > row) v = -1e30f;
        p[ct][i] = v;
      }
    }

    // Online softmax. Row r's 16 columns live in one 16-lane half -> xor 1..8.
#pragma unroll
    for (int i = 0; i < 8; ++i) {
      float mx = fmaxf(fmaxf(p[0][i], p[1][i]), fmaxf(p[2][i], p[3][i]));
#pragma unroll
      for (int msk = 8; msk >= 1; msk >>= 1)
        mx = fmaxf(mx, __shfl_xor(mx, msk, 32));
      float mn = fmaxf(m_[i], mx);
      float corr = __expf(m_[i] - mn);
      float rs = 0.f;
#pragma unroll
      for (int ct = 0; ct < 4; ++ct) {
        float pe = __expf(p[ct][i] - mn);
        rs += pe;
        Pl[w][i + 8 * half][ct * 16 + l15] = f2bf(pe);
      }
#pragma unroll
      for (int msk = 8; msk >= 1; msk >>= 1) rs += __shfl_xor(rs, msk, 32);
      l_[i] = l_[i] * corr + rs;
      m_[i] = mn;
#pragma unroll
      for (int dt = 0; dt < 4; ++dt) o[dt][i] *= corr;
    }
    __syncthreads();

    // O += P * V. P re-read in A-layout; V pre-transposed -> contiguous s loads.
    v16bf pf0 = frag_ld(&Pl[w][l15][0], half);    // s 0..31
    v16bf pf1 = frag_ld(&Pl[w][l15][32], half);   // s 32..63
#pragma unroll
    for (int dt = 0; dt < 4; ++dt) {
      const bf16_t* vrow = Vt + baseVt + (size_t)(dt * 16 + l15) * TT + s0;
      v16bf vf0 = frag_ld(vrow, half);
      v16bf vf1 = frag_ld(vrow + 32, half);
      o[dt] = wmma_bf16(pf0, vf0, o[dt]);
      o[dt] = wmma_bf16(pf1, vf1, o[dt]);
    }
    __syncthreads();
  }

  // Normalize and store bf16 [B,T,C] for the output projection.
  const int b = bh / HH, h = bh % HH;
#pragma unroll
  for (int dt = 0; dt < 4; ++dt)
#pragma unroll
    for (int i = 0; i < 8; ++i) {
      int t = trow0 + i + 8 * half;
      int c = h * DD + dt * 16 + l15;
      O[((size_t)(b * TT + t)) * CC + c] = f2bf(o[dt][i] / l_[i]);
    }
}

// ----------------------------------------------------------------- launch ---
extern "C" void kernel_launch(void* const* d_in, const int* in_sizes, int n_in,
                              void* d_out, int out_size, void* d_ws, size_t ws_size,
                              hipStream_t stream) {
  (void)in_sizes; (void)n_in; (void)out_size; (void)ws_size;
  const float* x  = (const float*)d_in[0];
  const float* Wq = (const float*)d_in[1];
  const float* Wk = (const float*)d_in[2];
  const float* Wv = (const float*)d_in[3];
  const float* Wo = (const float*)d_in[4];

  const size_t nX = (size_t)BB * TT * CC;  // 6,291,456
  const size_t nW = (size_t)CC * CC;       //   589,824

  bf16_t* xb  = (bf16_t*)d_ws;
  bf16_t* wqb = xb + nX;
  bf16_t* wkb = wqb + nW;
  bf16_t* wvb = wkb + nW;
  bf16_t* wob = wvb + nW;
  bf16_t* Qb  = wob + nW;   // [B,H,T,D]
  bf16_t* Kb  = Qb + nX;    // [B,H,T,D]
  bf16_t* Vtb = Kb + nX;    // [B,H,D,T]
  bf16_t* Ob  = Vtb + nX;   // [B,T,C]
  // total: 5*nX + 4*nW bf16 elements ~= 65.4 MB of d_ws

  cvt_f32_bf16<<<1024, 256, 0, stream>>>(x,  xb,  (int)nX);
  cvt_f32_bf16<<<512,  256, 0, stream>>>(Wq, wqb, (int)nW);
  cvt_f32_bf16<<<512,  256, 0, stream>>>(Wk, wkb, (int)nW);
  cvt_f32_bf16<<<512,  256, 0, stream>>>(Wv, wvb, (int)nW);
  cvt_f32_bf16<<<512,  256, 0, stream>>>(Wo, wob, (int)nW);

  dim3 gg(CC / 128, (BB * TT) / 128);  // (6, 64)
  gemm_bf16<0><<<gg, 256, 0, stream>>>(xb, wqb, Qb,  BB * TT, CC, CC);
  gemm_bf16<0><<<gg, 256, 0, stream>>>(xb, wkb, Kb,  BB * TT, CC, CC);
  gemm_bf16<1><<<gg, 256, 0, stream>>>(xb, wvb, Vtb, BB * TT, CC, CC);

  flash_attn<<<dim3(TT / 64, BB * HH), 128, 0, stream>>>(Qb, Kb, Vtb, Ob);

  gemm_bf16<2><<<gg, 256, 0, stream>>>(Ob, wob, d_out, BB * TT, CC, CC);
}
